// SingleHeadAttention_23304492548767
// MI455X (gfx1250) — matrix operand order repaired and checked by reading, not compiled, not run on hardware
//
#include <hip/hip_runtime.h>

typedef _Float16 half_t;
typedef __attribute__((ext_vector_type(16))) _Float16 v16h;
typedef __attribute__((ext_vector_type(8)))  _Float16 v8h;
typedef __attribute__((ext_vector_type(8)))  float    v8f;
typedef __attribute__((ext_vector_type(4)))  float    v4f;
typedef int b128_t __attribute__((vector_size(16)));   // payload type of *_b128 builtins

#define BATCH 8
#define SEQ   2048
#define EDIM  1024
#define HDIM  64
#define ROWS  (BATCH * SEQ)   // 16384
#define BK    64              // attention key-block size

#if __has_builtin(__builtin_amdgcn_sched_barrier)
#define SCHED_FENCE() __builtin_amdgcn_sched_barrier(0)
#else
#define SCHED_FENCE()
#endif

#if __has_builtin(__builtin_amdgcn_global_load_async_to_lds_b128) && \
    __has_builtin(__builtin_amdgcn_s_wait_asynccnt)
#define HAVE_ASYNC_LDS 1
#else
#define HAVE_ASYNC_LDS 0
#endif

// 16-byte global -> LDS copy: async DMA (ASYNCcnt) when available.
__device__ inline void copy16_g2l(const half_t* g, half_t* l) {
#if HAVE_ASYNC_LDS
    __builtin_amdgcn_global_load_async_to_lds_b128(
        (__attribute__((address_space(1))) b128_t*)g,
        (__attribute__((address_space(3))) b128_t*)l, 0, 0);
#else
    *(v8h*)l = *(const v8h*)g;
#endif
}
__device__ inline void async_wait0() {
#if HAVE_ASYNC_LDS
    __builtin_amdgcn_s_wait_asynccnt(0);
#endif
}

// ---- cross-lane helpers: xor-butterfly within each 16-lane half-group ----
template <int PAT>
__device__ inline float ds_swz(float v) {
    int i = __builtin_bit_cast(int, v);
    i = __builtin_amdgcn_ds_swizzle(i, PAT);
    return __builtin_bit_cast(float, i);
}
__device__ inline float rowmax16(float v) {
    v = fmaxf(v, ds_swz<(1 << 10) | 0x1f>(v));
    v = fmaxf(v, ds_swz<(2 << 10) | 0x1f>(v));
    v = fmaxf(v, ds_swz<(4 << 10) | 0x1f>(v));
    v = fmaxf(v, ds_swz<(8 << 10) | 0x1f>(v));
    return v;
}
__device__ inline float rowsum16(float v) {
    v += ds_swz<(1 << 10) | 0x1f>(v);
    v += ds_swz<(2 << 10) | 0x1f>(v);
    v += ds_swz<(4 << 10) | 0x1f>(v);
    v += ds_swz<(8 << 10) | 0x1f>(v);
    return v;
}

// =====================================================================
// Kernel 1: QKV projection.  out[r][h] = sum_e x[r][e] * W[e][h], f16 out.
// Q pre-scaled by 1/sqrt(H)=0.125.  Wave tile: 32 rows x 64 cols
// (2 A tiles x 4 B tiles -> 8 WMMAs per K-step).
// grid = (ROWS/256, 3), block = 256 (8 waves, 32 rows each).
// =====================================================================
__global__ __launch_bounds__(256) void qkv_proj_kernel(
    const float* __restrict__ x,
    const float* __restrict__ Wq, const float* __restrict__ Wk,
    const float* __restrict__ Wv,
    half_t* __restrict__ q16, half_t* __restrict__ k16,
    half_t* __restrict__ v16)
{
    const int which = blockIdx.y;
    const float* W = (which == 0) ? Wq : (which == 1) ? Wk : Wv;
    half_t* out = (which == 0) ? q16 : (which == 1) ? k16 : v16;
    const float oscale = (which == 0) ? 0.125f : 1.0f;

    // W slice transposed: Wt[n][k], k-slice of 128, stride padded to 136
    __shared__ alignas(16) half_t Wt[HDIM][136];

    const int tid  = threadIdx.x;
    const int wave = tid >> 5;
    const int lane = tid & 31;
    const int m0   = blockIdx.x * 256 + wave * 32;
    const int mrow = lane & 15;
    const int lo   = (lane < 16) ? 0 : 8;   // A-layout K-chunk base per half-wave
    const int half_off = (lane < 16) ? 0 : 8;

    v8f acc[2][4] = {};   // [m-tile][n-tile]

    for (int ks = 0; ks < EDIM; ks += 128) {
        __syncthreads();
        // stage W[ks:ks+128][0:64] transposed into LDS (f32 -> f16)
        for (int i = tid; i < 128 * HDIM / 4; i += 256) {
            const int k = (i * 4) / HDIM;
            const int n = (i * 4) % HDIM;
            const float4 w4 = *(const float4*)(W + (size_t)(ks + k) * HDIM + n);
            Wt[n + 0][k] = (half_t)w4.x;
            Wt[n + 1][k] = (half_t)w4.y;
            Wt[n + 2][k] = (half_t)w4.z;
            Wt[n + 3][k] = (half_t)w4.w;
        }
        __syncthreads();

        #pragma unroll
        for (int kk = 0; kk < 128; kk += 32) {
            // 2 A tiles: x[m0+m*16+mrow][ks+kk ..) in 16-bit A layout
            v16h a[2];
            #pragma unroll
            for (int m = 0; m < 2; ++m) {
                const float* xp =
                    x + (size_t)(m0 + m * 16 + mrow) * EDIM + ks + kk + lo;
                const v4f a0 = *(const v4f*)(xp);
                const v4f a1 = *(const v4f*)(xp + 4);
                const v4f a2 = *(const v4f*)(xp + 16);
                const v4f a3 = *(const v4f*)(xp + 20);
                #pragma unroll
                for (int j = 0; j < 4; ++j) {
                    a[m][j]      = (half_t)a0[j];
                    a[m][4 + j]  = (half_t)a1[j];
                    a[m][8 + j]  = (half_t)a2[j];
                    a[m][12 + j] = (half_t)a3[j];
                }
            }
            const int kbase = kk + ((lane < 16) ? 0 : 8);  // B-layout K base
            // preload ALL 4 B tiles, fence, then 8 independent WMMAs
            v16h bt[4];
            #pragma unroll
            for (int n = 0; n < 4; ++n) {
                const half_t* bp = &Wt[n * 16 + mrow][kbase];
                const v8h b0 = *(const v8h*)bp;
                const v8h b1 = *(const v8h*)(bp + 16);
                #pragma unroll
                for (int j = 0; j < 8; ++j) { bt[n][j] = b0[j]; bt[n][8 + j] = b1[j]; }
            }
            SCHED_FENCE();
            #pragma unroll
            for (int n = 0; n < 4; ++n)
                #pragma unroll
                for (int m = 0; m < 2; ++m)
                    acc[m][n] = __builtin_amdgcn_wmma_f32_16x16x32_f16(
                        false, a[m], false, bt[n], (short)0, acc[m][n], false, false);
            SCHED_FENCE();
        }
    }

    // write f16 outputs (C layout: VGPR r -> M = r or r+8 by half-wave)
    #pragma unroll
    for (int m = 0; m < 2; ++m)
        #pragma unroll
        for (int n = 0; n < 4; ++n)
            #pragma unroll
            for (int r = 0; r < 8; ++r) {
                const int rr = m0 + m * 16 + r + half_off;
                out[(size_t)rr * HDIM + n * 16 + mrow] =
                    (half_t)(acc[m][n][r] * oscale);
            }
}

// =====================================================================
// Kernel 2: causal flash attention over f16 Q/K/V, f32 out.
// 64-key blocks; K staged by async global->LDS DMA (ASYNCcnt),
// V staged via VGPRs (transposed on store); double-buffered.
// Protocol: async for block kb+1 issued AFTER barrier kb (old readers
// of that buffer are provably done); each wave s_wait_asynccnt(0)
// BEFORE barrier kb so the barrier publishes completed DMA to all waves.
// grid = (SEQ/64, BATCH), block = 128 (4 waves, 16 query rows each).
// =====================================================================
__global__ __launch_bounds__(128) void flash_attn_kernel(
    const half_t* __restrict__ q16, const half_t* __restrict__ k16,
    const half_t* __restrict__ v16, float* __restrict__ out)
{
    const int b    = blockIdx.y;
    const int q0   = blockIdx.x * 64;
    const int tid  = threadIdx.x;
    const int wave = tid >> 5;
    const int lane = tid & 31;
    const int m0   = q0 + wave * 16;
    const int mrow = lane & 15;
    const int lo   = (lane < 16) ? 0 : 8;
    const int bko  = (lane < 16) ? 0 : 8;   // B-layout K base

    __shared__ alignas(16) half_t Klds[2][BK][72];    // [buf][key][h]
    __shared__ alignas(16) half_t Vt[2][HDIM][72];    // [buf][h][key]
    __shared__ alignas(16) half_t Plds[4][16][72];    // per-wave P scratch

    const half_t* qb = q16 + (size_t)b * SEQ * HDIM;
    const half_t* kg = k16 + (size_t)b * SEQ * HDIM;
    const half_t* vg = v16 + (size_t)b * SEQ * HDIM;

    // Q in A layout: two K-steps over H=64 (Q already scaled by 0.125)
    v16h aq[2];
    #pragma unroll
    for (int kh = 0; kh < 2; ++kh) {
        const half_t* qp = qb + (size_t)(m0 + mrow) * HDIM + kh * 32 + lo;
        const v8h c0 = *(const v8h*)qp;
        const v8h c1 = *(const v8h*)(qp + 16);
        #pragma unroll
        for (int j = 0; j < 8; ++j) { aq[kh][j] = c0[j]; aq[kh][8 + j] = c1[j]; }
    }

    v8f o[4] = {};
    float mrun[8], lrun[8];
    #pragma unroll
    for (int r = 0; r < 8; ++r) { mrun[r] = -1e30f; lrun[r] = 0.0f; }

    // staging assignment: thread -> 4 chunks of 8 f16 per array,
    // rows r0+{0,16,32,48} at fixed column.
    const int r0  = tid >> 3;
    const int col = (tid & 7) * 8;

    const int nkb = blockIdx.x + 1;   // causal: key blocks up to the diagonal
    v8h vr[4];

    // prologue: async K block 0 -> buf0; V block 0 -> regs
    #pragma unroll
    for (int i = 0; i < 4; ++i) {
        vr[i] = *(const v8h*)(vg + (size_t)(r0 + 16 * i) * HDIM + col);
        copy16_g2l(kg + (size_t)(r0 + 16 * i) * HDIM + col,
                   &Klds[0][r0 + 16 * i][col]);
    }

    for (int kb = 0; kb < nkb; ++kb) {
        const int buf = kb & 1;
        // ---- V transpose-store into LDS buffer ----
        #pragma unroll
        for (int i = 0; i < 4; ++i)
            #pragma unroll
            for (int j = 0; j < 8; ++j) Vt[buf][col + j][r0 + 16 * i] = vr[i][j];
        // ---- next block's V into regs (latency hidden by compute) ----
        if (kb + 1 < nkb) {
            const int key1 = (kb + 1) * BK;
            #pragma unroll
            for (int i = 0; i < 4; ++i)
                vr[i] = *(const v8h*)(vg + (size_t)(key1 + r0 + 16 * i) * HDIM + col);
        }
        // ---- prefetch block kb+2 into cache ----
        if (kb + 2 < nkb) {
            const int key2 = (kb + 2) * BK;
            #pragma unroll
            for (int i = 0; i < 4; ++i) {
                __builtin_prefetch(kg + (size_t)(key2 + r0 + 16 * i) * HDIM + col, 0, 1);
                __builtin_prefetch(vg + (size_t)(key2 + r0 + 16 * i) * HDIM + col, 0, 1);
            }
        }
        async_wait0();      // this wave's K DMA for block kb is in LDS
        __syncthreads();    // publish everyone's DMA + Vt stores
        // ---- async K DMA for block kb+1 into the other buffer ----
        if (kb + 1 < nkb) {
            const int key1 = (kb + 1) * BK;
            #pragma unroll
            for (int i = 0; i < 4; ++i)
                copy16_g2l(kg + (size_t)(key1 + r0 + 16 * i) * HDIM + col,
                           &Klds[buf ^ 1][r0 + 16 * i][col]);
        }

        const int key0 = kb * BK;

        // ---- scores: S[16q x 64k] via 8 WMMAs (kh-major: independent accums) ----
        v8f s[4] = {};
        #pragma unroll
        for (int kh = 0; kh < 2; ++kh) {
            v16h bt[4];
            #pragma unroll
            for (int t = 0; t < 4; ++t) {
                const half_t* bp = &Klds[buf][t * 16 + mrow][kh * 32 + bko];
                const v8h b0 = *(const v8h*)bp;
                const v8h b1 = *(const v8h*)(bp + 16);
                #pragma unroll
                for (int j = 0; j < 8; ++j) { bt[t][j] = b0[j]; bt[t][8 + j] = b1[j]; }
            }
            SCHED_FENCE();
            #pragma unroll
            for (int t = 0; t < 4; ++t)
                s[t] = __builtin_amdgcn_wmma_f32_16x16x32_f16(
                    false, aq[kh], false, bt[t], (short)0, s[t], false, false);
            SCHED_FENCE();
        }

        // ---- causal mask + online softmax (C layout) ----
        float alpha[8];
        #pragma unroll
        for (int r = 0; r < 8; ++r) {
            const int qrow = m0 + r + ((lane < 16) ? 0 : 8);
            float sv[4];
            #pragma unroll
            for (int t = 0; t < 4; ++t)
                sv[t] = (key0 + t * 16 + mrow <= qrow) ? s[t][r] : -1e30f;
            const float mx = rowmax16(fmaxf(fmaxf(sv[0], sv[1]), fmaxf(sv[2], sv[3])));
            const float mnew = fmaxf(mrun[r], mx);
            alpha[r] = __expf(mrun[r] - mnew);
            mrun[r]  = mnew;
            float psum = 0.0f;
            const int rr = r + ((lane < 16) ? 0 : 8);
            #pragma unroll
            for (int t = 0; t < 4; ++t) {
                const float p = __expf(sv[t] - mnew);
                psum += p;
                Plds[wave][rr][t * 16 + mrow] = (half_t)p;
            }
            lrun[r] = lrun[r] * alpha[r] + rowsum16(psum);
        }

        // rescale running output
        #pragma unroll
        for (int n = 0; n < 4; ++n)
            #pragma unroll
            for (int r = 0; r < 8; ++r) o[n][r] *= alpha[r];

        // reload P in A layout: two K-steps over 64 keys
        v16h ap[2];
        #pragma unroll
        for (int t2 = 0; t2 < 2; ++t2) {
            const half_t* pp = &Plds[wave][mrow][t2 * 32 + lo];
            const v8h p0v = *(const v8h*)pp;
            const v8h p1v = *(const v8h*)(pp + 16);
            #pragma unroll
            for (int j = 0; j < 8; ++j) { ap[t2][j] = p0v[j]; ap[t2][8 + j] = p1v[j]; }
        }

        // ---- O += P x V : 8 WMMAs (t2-major: independent accums inside) ----
        #pragma unroll
        for (int t2 = 0; t2 < 2; ++t2) {
            v16h bv[4];
            #pragma unroll
            for (int n = 0; n < 4; ++n) {
                const half_t* bp = &Vt[buf][n * 16 + mrow][t2 * 32 + bko];
                const v8h b0 = *(const v8h*)bp;
                const v8h b1 = *(const v8h*)(bp + 16);
                #pragma unroll
                for (int j = 0; j < 8; ++j) { bv[n][j] = b0[j]; bv[n][8 + j] = b1[j]; }
            }
            SCHED_FENCE();
            #pragma unroll
            for (int n = 0; n < 4; ++n)
                o[n] = __builtin_amdgcn_wmma_f32_16x16x32_f16(
                    false, ap[t2], false, bv[n], (short)0, o[n], false, false);
            SCHED_FENCE();
        }
    }

    // normalize and store f32
    float rl[8];
    #pragma unroll
    for (int r = 0; r < 8; ++r) rl[r] = 1.0f / lrun[r];
    #pragma unroll
    for (int n = 0; n < 4; ++n) {
        #pragma unroll
        for (int r = 0; r < 8; ++r) {
            const int qrow = m0 + r + ((lane < 16) ? 0 : 8);
            out[((size_t)b * SEQ + qrow) * HDIM + n * 16 + mrow] = o[n][r] * rl[r];
        }
    }
}

extern "C" void kernel_launch(void* const* d_in, const int* in_sizes, int n_in,
                              void* d_out, int out_size, void* d_ws, size_t ws_size,
                              hipStream_t stream) {
    const float* x  = (const float*)d_in[0];
    const float* Wq = (const float*)d_in[1];
    const float* Wk = (const float*)d_in[2];
    const float* Wv = (const float*)d_in[3];
    float* out = (float*)d_out;

    // workspace: f16 Q, K, V  (3 x 16384 x 64 x 2B = 6 MB)
    half_t* q16 = (half_t*)d_ws;
    half_t* k16 = q16 + (size_t)ROWS * HDIM;
    half_t* v16 = k16 + (size_t)ROWS * HDIM;

    dim3 g1(ROWS / 256, 3);
    qkv_proj_kernel<<<g1, 256, 0, stream>>>(x, Wq, Wk, Wv, q16, k16, v16);

    dim3 g2(SEQ / 64, BATCH);
    flash_attn_kernel<<<g2, 128, 0, stream>>>(q16, k16, v16, out);
}
